// VectorQuantizer_33646773797150
// MI455X (gfx1250) — matrix operand order repaired and checked by reading, not compiled, not run on hardware
//
#include <hip/hip_runtime.h>

// ---------------------------------------------------------------------------
// VQ (vector quantizer) for MI455X / gfx1250.
// B=32768 rows of z (D=512) vs K=4096 codes. Distance GEMM fused with argmin:
//   - v_wmma_f32_16x16x32_bf16, A tile resident in 128 VGPRs per wave
//   - B slabs staged into double-buffered LDS via GLOBAL_LOAD_ASYNC_TO_LDS_B128
//     (ASYNCcnt pipeline, one chunk ahead), sync-copy fallback if the builtin
//     is unavailable on this toolchain.
// ---------------------------------------------------------------------------

typedef __bf16 bf16_t;
typedef __attribute__((ext_vector_type(16))) __bf16 v16bf;
typedef __attribute__((ext_vector_type(8)))  float  v8f;
typedef __attribute__((ext_vector_type(4)))  float  v4f;
typedef __attribute__((ext_vector_type(4)))  int    v4i;

// address-space-qualified pointer types for the async-to-LDS builtin
// (param 0 is AS1 int4*, param 1 is AS3 int4*, per hipcc's diagnostic)
typedef __attribute__((address_space(1))) v4i as1_v4i;
typedef __attribute__((address_space(3))) v4i as3_v4i;

#define NB    32768   // batch rows
#define DDIM  512     // feature dim
#define KCODE 4096    // codebook entries
#define BETA  0.25f

#if __has_builtin(__builtin_amdgcn_global_load_async_to_lds_b128)
#define VQ_ASYNC 1
#else
#define VQ_ASYNC 0
#endif

#if VQ_ASYNC
#if __has_builtin(__builtin_amdgcn_s_wait_asynccnt)
#define VQ_WAIT_ASYNC(n) __builtin_amdgcn_s_wait_asynccnt(n)
#else
#define VQ_WAIT_ASYNC(n) asm volatile("s_wait_asynccnt %0" ::"i"(n) : "memory")
#endif
#else
#define VQ_WAIT_ASYNC(n)
#endif

// workspace layout (bytes)
#define WS_EMB_BF16   0u                            // KCODE*DDIM*2 = 4 MiB
#define WS_E2         (KCODE * DDIM * 2u)           // KCODE * 4
#define WS_IDX        (WS_E2 + KCODE * 4u)          // NB * 4 (int)
#define WS_PARTIAL    (WS_IDX + (unsigned)NB * 4u)  // NB * 4 (float)

// ---------------------------------------------------------------------------
// Kernel 1: emb f32 -> bf16 (row major) + e2[k] = ||emb[k]||^2 in f32.
// ---------------------------------------------------------------------------
__global__ void vq_prep(const float* __restrict__ emb,
                        bf16_t* __restrict__ emb_bf,
                        float* __restrict__ e2) {
  const int k = blockIdx.x;
  const float* row = emb + (size_t)k * DDIM;
  bf16_t* orow = emb_bf + (size_t)k * DDIM;
  float s = 0.f;
  for (int d = threadIdx.x; d < DDIM; d += 256) {
    float v = row[d];
    orow[d] = (bf16_t)v;   // RNE f32->bf16
    s += v * v;
  }
  __shared__ float red[8];
  for (int m = 16; m >= 1; m >>= 1) s += __shfl_xor(s, m, 32);
  const int lane = threadIdx.x & 31, wv = threadIdx.x >> 5;
  if (lane == 0) red[wv] = s;
  __syncthreads();
  if (threadIdx.x == 0) {
    float t = 0.f;
    for (int i = 0; i < 8; ++i) t += red[i];
    e2[k] = t;
  }
}

// ---------------------------------------------------------------------------
// Kernel 2: fused distance GEMM + argmin.
// Grid: NB/128 blocks, 256 threads (8 waves). Wave w owns rows
// blockIdx.x*128 + w*16 .. +15. A tile (16 x 512 bf16) resident in 128 VGPRs.
// 256 chunks of 16 codes; double-buffered 16 KB B slabs in LDS, prefetched
// one chunk ahead with async-to-LDS; 16 chained WMMAs per chunk.
// ---------------------------------------------------------------------------
__launch_bounds__(256, 1)
__global__ void vq_argmin(const float* __restrict__ z,
                          const bf16_t* __restrict__ emb_bf,
                          const float* __restrict__ e2,
                          int* __restrict__ idx_i,
                          float* __restrict__ idx_f) {
  // B slab, fragment-swizzled: chunk t (t=d/32) holds 32(K) x 16(N) halves,
  // half index = t*512 + lane*16 + j, with lane = N + 16*(K in hi half).
  __shared__ __align__(32) bf16_t lds_b[2][16 * 512];

  const int lane = threadIdx.x & 31;
  const int wv   = threadIdx.x >> 5;
  const int hi   = lane >> 4;          // 0: lanes 0-15, 1: lanes 16-31
  const int col  = lane & 15;          // N column owned by this lane (C layout)

  // ---- load + convert resident A tile (z rows, f32 -> bf16) ----
  // A layout (16-bit 16x32): row M = lane&15; half j -> K = j + 8*(j>=8) + 8*hi
  const int mrow = blockIdx.x * 128 + wv * 16 + (lane & 15);
  const float* zrow = z + (size_t)mrow * DDIM;
  v16bf a[16];
#pragma unroll
  for (int c = 0; c < 16; ++c) {
    const int d0 = c * 32 + hi * 8;
    v4f f0 = *(const v4f*)(zrow + d0);         // j 0..3
    v4f f1 = *(const v4f*)(zrow + d0 + 4);     // j 4..7
    v4f f2 = *(const v4f*)(zrow + d0 + 16);    // j 8..11
    v4f f3 = *(const v4f*)(zrow + d0 + 20);    // j 12..15
#pragma unroll
    for (int q = 0; q < 4; ++q) {
      a[c][q]      = (bf16_t)f0[q];
      a[c][q + 4]  = (bf16_t)f1[q];
      a[c][q + 8]  = (bf16_t)f2[q];
      a[c][q + 12] = (bf16_t)f3[q];
    }
  }

  // stage one 16-code slab into lds_b[buf] (async if available, else sync)
  auto stage = [&](int nb, int buf) {
    const bf16_t* ebase = emb_bf + (size_t)nb * DDIM;
#pragma unroll
    for (int it = 0; it < 4; ++it) {
      const int gid = threadIdx.x + 256 * it;  // 1024 groups of 8 halves
      const int n   = gid >> 6;                // code within chunk
      const int d0  = (gid & 63) << 3;         // 8 consecutive d
      const int t   = d0 >> 5;
      const int bl  = n + (((d0 >> 3) & 1) << 4);
      const int jb  = ((d0 >> 4) & 1) << 3;    // 0 or 8
      bf16_t* src = const_cast<bf16_t*>(ebase + (size_t)n * DDIM + d0);
      bf16_t* dst = &lds_b[buf][t * 512 + bl * 16 + jb];
#if VQ_ASYNC
      __builtin_amdgcn_global_load_async_to_lds_b128(
          (as1_v4i*)src, (as3_v4i*)dst, 0, 0);
#else
      *(uint4*)dst = *(const uint4*)src;
#endif
    }
  };

  float minv[8];
  int   mini[8];
#pragma unroll
  for (int r = 0; r < 8; ++r) { minv[r] = 3.4e38f; mini[r] = 0; }

  stage(0, 0);  // prologue prefetch: chunk 0 -> buf 0 (4 async ops in flight)

  for (int i = 0; i < KCODE / 16; ++i) {
    const int nb  = i * 16;
    const int cur = i & 1;

    // all waves done reading buf cur^1 -> safe to overwrite it
    __syncthreads();
    if (nb + 16 < KCODE) {
      stage(nb + 16, cur ^ 1);   // prefetch next chunk (4 newer async ops)
      VQ_WAIT_ASYNC(4);          // older 4 (buf cur) complete, newer in flight
    } else {
      VQ_WAIT_ASYNC(0);          // drain for the final chunk
    }
    __syncthreads();             // publish buf cur to all 8 waves

    // ---- 16 chained WMMAs over D=512 ----
    v8f acc = {0.f, 0.f, 0.f, 0.f, 0.f, 0.f, 0.f, 0.f};
#pragma unroll
    for (int t = 0; t < 16; ++t) {
      v16bf bfrag = *(const v16bf*)&lds_b[cur][t * 512 + lane * 16];
      acc = __builtin_amdgcn_wmma_f32_16x16x32_bf16(
          false, a[t], false, bfrag, (short)0, acc, false, false);
    }

    // ---- argmin epilogue: dist' = e2[n] - 2*(z.e) (z^2 is row-constant) ----
    const float e2v = e2[nb + col];
#pragma unroll
    for (int r = 0; r < 8; ++r) {
      const float cand = fmaf(-2.f, acc[r], e2v);
      if (cand < minv[r]) { minv[r] = cand; mini[r] = nb + col; }
    }
  }

  // ---- reduce over the 16 lanes sharing each row (xor within half-wave) ----
#pragma unroll
  for (int m = 1; m <= 8; m <<= 1) {
#pragma unroll
    for (int r = 0; r < 8; ++r) {
      const float ov = __shfl_xor(minv[r], m, 32);
      const int   oi = __shfl_xor(mini[r], m, 32);
      if (ov < minv[r] || (ov == minv[r] && oi < mini[r])) {
        minv[r] = ov; mini[r] = oi;
      }
    }
  }
  if (col == 0) {  // lanes 0 (rows r) and 16 (rows r+8)
    const int rowbase = blockIdx.x * 128 + wv * 16 + hi * 8;
#pragma unroll
    for (int r = 0; r < 8; ++r) {
      idx_i[rowbase + r] = mini[r];
      idx_f[rowbase + r] = (float)mini[r];
    }
  }
}

// ---------------------------------------------------------------------------
// Kernel 3: gather z_q = emb[idx] (f32) -> d_out, per-row Sum((z-q)^2).
// ---------------------------------------------------------------------------
__global__ void vq_gather(const float* __restrict__ z,
                          const float* __restrict__ emb,
                          const int* __restrict__ idx_i,
                          float* __restrict__ zq_out,
                          float* __restrict__ partials) {
  const int row = blockIdx.x;
  const int k = idx_i[row];
  const float* q  = emb + (size_t)k * DDIM;
  const float* zr = z + (size_t)row * DDIM;
  float s = 0.f;
  for (int d = threadIdx.x; d < DDIM; d += 256) {
    const float qv = q[d];
    const float df = zr[d] - qv;
    zq_out[(size_t)row * DDIM + d] = qv;
    s += df * df;
  }
  __shared__ float red[8];
  for (int m = 16; m >= 1; m >>= 1) s += __shfl_xor(s, m, 32);
  const int lane = threadIdx.x & 31, wv = threadIdx.x >> 5;
  if (lane == 0) red[wv] = s;
  __syncthreads();
  if (threadIdx.x == 0) {
    float t = 0.f;
    for (int i = 0; i < 8; ++i) t += red[i];
    partials[row] = t;
  }
}

// ---------------------------------------------------------------------------
// Kernel 4: finalize vq_loss = (1+BETA) * sum(partials) / (B*D).
// ---------------------------------------------------------------------------
__global__ void vq_loss_fin(const float* __restrict__ partials,
                            float* __restrict__ out_loss) {
  float s = 0.f;
  for (int i = threadIdx.x; i < NB; i += 256) s += partials[i];
  __shared__ float red[8];
  for (int m = 16; m >= 1; m >>= 1) s += __shfl_xor(s, m, 32);
  const int lane = threadIdx.x & 31, wv = threadIdx.x >> 5;
  if (lane == 0) red[wv] = s;
  __syncthreads();
  if (threadIdx.x == 0) {
    float t = 0.f;
    for (int i = 0; i < 8; ++i) t += red[i];
    *out_loss = (1.0f + BETA) * t / (float)((size_t)NB * DDIM);
  }
}

// ---------------------------------------------------------------------------
extern "C" void kernel_launch(void* const* d_in, const int* in_sizes, int n_in,
                              void* d_out, int out_size, void* d_ws, size_t ws_size,
                              hipStream_t stream) {
  (void)in_sizes; (void)n_in; (void)out_size; (void)ws_size;
  const float* z   = (const float*)d_in[0];   // (B, D) f32
  const float* emb = (const float*)d_in[1];   // (K, D) f32

  char* ws = (char*)d_ws;
  bf16_t* emb_bf   = (bf16_t*)(ws + WS_EMB_BF16);
  float*  e2       = (float*)(ws + WS_E2);
  int*    idx_i    = (int*)(ws + WS_IDX);
  float*  partials = (float*)(ws + WS_PARTIAL);

  float* out    = (float*)d_out;              // z_q_st : B*D
  float* idx_f  = out + (size_t)NB * DDIM;    // idx    : B (as f32)
  float* loss_p = idx_f + NB;                 // vq_loss: 1

  vq_prep<<<KCODE, 256, 0, stream>>>(emb, emb_bf, e2);
  vq_argmin<<<NB / 128, 256, 0, stream>>>(z, emb_bf, e2, idx_i, idx_f);
  vq_gather<<<NB, 256, 0, stream>>>(z, emb, idx_i, out, partials);
  vq_loss_fin<<<1, 256, 0, stream>>>(partials, loss_p);
}
